// RelativeMultiHeadAttn_54047868453499
// MI455X (gfx1250) — matrix operand order repaired and checked by reading, not compiled, not run on hardware
//
#include <hip/hip_runtime.h>
#include <hip/hip_bf16.h>

// ---------------------------------------------------------------------------
// Transformer-XL relative multi-head attention, MI455X (gfx1250, wave32).
// All GEMMs via v_wmma_f32_16x16x32_bf16 (fp32 accum). BD term collapses to
// rank-1 (r = arange): BD[i,j] = (j<=i) ? r[j-i+Q-1]*s[i,b,h] : 0.
// Memory-bound: 822MB of mandatory tuple output writes dominate (~35us @23.3TB/s).
// V^T operand tiles in the P.V GEMM are streamed VMEM->LDS with
// global_load_async_to_lds_b128 (ASYNCcnt, double-buffered).
// ---------------------------------------------------------------------------

typedef __bf16 bf16_t;
typedef __attribute__((ext_vector_type(16))) __bf16 v16bf;
typedef __attribute__((ext_vector_type(8)))  float  v8f;

union Abf { uint4 u[2]; v16bf v; };

static __device__ __forceinline__ v8f zero8() {
  v8f z = {0.f, 0.f, 0.f, 0.f, 0.f, 0.f, 0.f, 0.f};
  return z;
}

static __device__ __forceinline__ v8f wmma_bf16(const Abf& a, const Abf& b, v8f c) {
  return __builtin_amdgcn_wmma_f32_16x16x32_bf16(false, a.v, false, b.v,
                                                 (short)0, c, false, false);
}

static __device__ __forceinline__ bf16_t f2b(float f) { return (bf16_t)f; }
static __device__ __forceinline__ float  b2f(bf16_t b) { return (float)b; }

#define QLEN   1024
#define BSZ    8
#define NHEAD  8
#define DHEAD  64
#define DMODEL 512
#define ROWS   (QLEN * BSZ)        // 8192
#define HDIM   (3 * NHEAD * DHEAD) // 1536
#define SCALE  (1.0f / 4096.0f)    // 1/d_head^2 as written in the module

// ---------------------------------------------------------------------------
// K0: fp32 -> bf16 staging of w (activations), qkv_W, o_W
// ---------------------------------------------------------------------------
__global__ void k_convert(const float* __restrict__ w,
                          const float* __restrict__ qkvW,
                          const float* __restrict__ oW,
                          bf16_t* __restrict__ Xbf,
                          bf16_t* __restrict__ Wbf,
                          bf16_t* __restrict__ oWbf) {
  const int nX = ROWS * DMODEL;      // 4194304
  const int nW = HDIM * DMODEL;      // 786432
  const int nO = DMODEL * DMODEL;    // 262144
  int t = blockIdx.x * blockDim.x + threadIdx.x;
  if (t < nX)                 Xbf [t]           = f2b(w   [t]);
  else if (t < nX + nW)       Wbf [t - nX]      = f2b(qkvW[t - nX]);
  else if (t < nX + nW + nO)  oWbf[t - nX - nW] = f2b(oW  [t - nX - nW]);
}

// ---------------------------------------------------------------------------
// K1: QKV projection  [8192 x 512] x [512 x 1536]^T, one wave per 16x16 tile.
// Epilogue scatters into staged Qw (q + r_w_bias), K, V^T (all bf16).
// ---------------------------------------------------------------------------
__global__ __launch_bounds__(32) void k_qkv_gemm(const bf16_t* __restrict__ X,
                                                 const bf16_t* __restrict__ W,
                                                 const float*  __restrict__ rwb,
                                                 bf16_t* __restrict__ Qw,
                                                 bf16_t* __restrict__ Kst,
                                                 bf16_t* __restrict__ Vt) {
  const int ti = blockIdx.x;             // 0..511 over 8192 rows
  const int tj = blockIdx.y;             // 0..95  over 1536 cols
  const int lane = threadIdx.x, lo = lane & 15, hi = lane >> 4;

  const bf16_t* arow = X + (size_t)(ti * 16 + lo) * DMODEL;
  const bf16_t* brow = W + (size_t)(tj * 16 + lo) * DMODEL;

  v8f c = zero8();
  for (int k0 = 0; k0 < DMODEL; k0 += 32) {
    Abf a, b;
    a.u[0] = *(const uint4*)(arow + k0 + 16 * hi);
    a.u[1] = *(const uint4*)(arow + k0 + 16 * hi + 8);
    b.u[0] = *(const uint4*)(brow + k0 + 16 * hi);
    b.u[1] = *(const uint4*)(brow + k0 + 16 * hi + 8);
    c = wmma_bf16(a, b, c);
  }

  const int gcol  = tj * 16 + lo;
  const int third = gcol >> 9;
  const int rem   = gcol & 511;
  const int h     = rem >> 6;
  const int d     = rem & 63;
  for (int r = 0; r < 8; ++r) {
    const int grow = ti * 16 + r + 8 * hi;   // row = i*BSZ + b
    const int i = grow >> 3, b_ = grow & 7;
    const float val = c[r];
    if (third == 0)      Qw [((size_t)(b_ * 8 + h) * QLEN + i) * DHEAD + d] = f2b(val + rwb[h * DHEAD + d]);
    else if (third == 1) Kst[((size_t)(b_ * 8 + h) * QLEN + i) * DHEAD + d] = f2b(val);
    else                 Vt [((size_t)(b_ * 8 + h) * DHEAD + d) * QLEN + i] = f2b(val);
  }
}

// ---------------------------------------------------------------------------
// K2: s[i,b,h] = (q + r_r_bias) . r_W_head  (rank-1 BD coefficient)
// ---------------------------------------------------------------------------
__global__ void k_s(const bf16_t* __restrict__ Qw,
                    const float* __restrict__ rW,
                    const float* __restrict__ rrb,
                    const float* __restrict__ rwb,
                    float* __restrict__ s) {
  int t = blockIdx.x * blockDim.x + threadIdx.x;   // 65536 = 8192*8
  if (t >= ROWS * NHEAD) return;
  const int h = t & 7, ib = t >> 3;
  const int i = ib >> 3, b_ = ib & 7;
  const bf16_t* q = Qw + ((size_t)(b_ * 8 + h) * QLEN + i) * DHEAD;
  float acc = 0.f;
  for (int d = 0; d < DHEAD; ++d) {
    const float rv = rW[h * DHEAD + d];
    acc += b2f(q[d]) * rv + (rrb[h * DHEAD + d] - rwb[h * DHEAD + d]) * rv;
  }
  s[t] = acc;
}

// ---------------------------------------------------------------------------
// K3: AC = Q.K^T per (b,h,i-tile) via WMMA; analytic BD; write AC & BD to
// d_out; online softmax stats (m, l) per row with cross-lane xor reduce.
// ---------------------------------------------------------------------------
__global__ __launch_bounds__(32) void k_score(const bf16_t* __restrict__ Qw,
                                              const bf16_t* __restrict__ Kst,
                                              const float*  __restrict__ rvec,
                                              const float*  __restrict__ s,
                                              float* __restrict__ outAC,
                                              float* __restrict__ outBD,
                                              float* __restrict__ mrow,
                                              float* __restrict__ lrow) {
  const int it = blockIdx.x;     // i-tile 0..63
  const int bh = blockIdx.y;     // b*8+h 0..63
  const int b_ = bh >> 3, h = bh & 7;
  const int lane = threadIdx.x, lo = lane & 15, hi = lane >> 4;

  const bf16_t* qbase = Qw  + (size_t)bh * QLEN * DHEAD;
  const bf16_t* kbase = Kst + (size_t)bh * QLEN * DHEAD;
  const bf16_t* qrow  = qbase + (size_t)(it * 16 + lo) * DHEAD;

  Abf a0, a1;
  a0.u[0] = *(const uint4*)(qrow + 16 * hi);
  a0.u[1] = *(const uint4*)(qrow + 16 * hi + 8);
  a1.u[0] = *(const uint4*)(qrow + 32 + 16 * hi);
  a1.u[1] = *(const uint4*)(qrow + 32 + 16 * hi + 8);

  float srow[8], m[8], l[8];
  for (int r = 0; r < 8; ++r) {
    const int i = it * 16 + r + 8 * hi;
    srow[r] = s[(size_t)(i * 8 + b_) * 8 + h];
    m[r] = -__builtin_inff();
    l[r] = 0.f;
  }

  for (int jt = 0; jt < 64; ++jt) {
    const bf16_t* krow = kbase + (size_t)(jt * 16 + lo) * DHEAD;
    if (jt < 63) __builtin_prefetch(krow + DHEAD * 16, 0, 1);
    Abf b0, b1;
    b0.u[0] = *(const uint4*)(krow + 16 * hi);
    b0.u[1] = *(const uint4*)(krow + 16 * hi + 8);
    b1.u[0] = *(const uint4*)(krow + 32 + 16 * hi);
    b1.u[1] = *(const uint4*)(krow + 32 + 16 * hi + 8);
    v8f c = zero8();
    c = wmma_bf16(a0, b0, c);
    c = wmma_bf16(a1, b1, c);

    const int j = jt * 16 + lo;
    for (int r = 0; r < 8; ++r) {
      const int i  = it * 16 + r + 8 * hi;
      const float ac = c[r];
      const float bd = (j <= i) ? rvec[j - i + (QLEN - 1)] * srow[r] : 0.f;
      const size_t o = ((size_t)i * QLEN + j) * (BSZ * NHEAD) + bh;
      outAC[o] = ac;
      outBD[o] = bd;
      const float sc = (ac + bd) * SCALE;
      const float mn = fmaxf(m[r], sc);
      l[r] = l[r] * __expf(m[r] - mn) + __expf(sc - mn);
      m[r] = mn;
    }
  }

  // combine 16 key-columns per lane-group (lanes 0-15 hold rows 0-7, 16-31 rows 8-15)
  for (int r = 0; r < 8; ++r) {
    for (int mask = 1; mask < 16; mask <<= 1) {
      const float mo = __shfl_xor(m[r], mask, 32);
      const float lx = __shfl_xor(l[r], mask, 32);
      const float mn = fmaxf(m[r], mo);
      l[r] = l[r] * __expf(m[r] - mn) + lx * __expf(mo - mn);
      m[r] = mn;
    }
    if (lo == 0) {
      const int i = it * 16 + r + 8 * hi;
      mrow[(size_t)bh * QLEN + i] = m[r];
      lrow[(size_t)bh * QLEN + i] = l[r];
    }
  }
}

// ---------------------------------------------------------------------------
// K4: recompute scores, prob = exp(sc-m)/l -> d_out; bf16 prob through LDS
// (layout swap C->A operand); V^T chunks streamed to LDS with async-to-LDS
// B128 (double-buffered, ASYNCcnt); WMMA prob x V^T -> attn_vec (bf16).
// ---------------------------------------------------------------------------
__global__ __launch_bounds__(32) void k_prob_av(const bf16_t* __restrict__ Qw,
                                                const bf16_t* __restrict__ Kst,
                                                const bf16_t* __restrict__ Vt,
                                                const float*  __restrict__ rvec,
                                                const float*  __restrict__ s,
                                                const float*  __restrict__ mrow,
                                                const float*  __restrict__ lrow,
                                                float* __restrict__ outP,
                                                bf16_t* __restrict__ AV) {
  __shared__ bf16_t ldsP[16 * 40];        // prob tile, stride 40 (bank-safe)
  __shared__ bf16_t ldsV[2][64 * 40];     // double-buffered V^T chunk 64x32

  const int it = blockIdx.x;
  const int bh = blockIdx.y;
  const int b_ = bh >> 3, h = bh & 7;
  const int lane = threadIdx.x, lo = lane & 15, hi = lane >> 4;

  const bf16_t* qbase = Qw  + (size_t)bh * QLEN * DHEAD;
  const bf16_t* kbase = Kst + (size_t)bh * QLEN * DHEAD;
  const bf16_t* vbase = Vt  + (size_t)bh * DHEAD * QLEN;
  const bf16_t* qrow  = qbase + (size_t)(it * 16 + lo) * DHEAD;

  // async VMEM->LDS staging of one 64(d) x 32(j) V^T chunk: 8 x b128 per wave
  const int vrsub = lane >> 2;   // 0..7  (row sub-index)
  const int vcc   = lane & 3;    // 0..3  (8-element column chunk)
  auto issue_async_v = [&](int j0, int buf) {
    for (int t = 0; t < 8; ++t) {
      const int row = t * 8 + vrsub;                       // d = 0..63
      unsigned int loff = (unsigned int)(uintptr_t)&ldsV[buf][row * 40 + vcc * 8];
      unsigned long long ga =
          (unsigned long long)(uintptr_t)(vbase + (size_t)row * QLEN + j0 + vcc * 8);
      asm volatile("global_load_async_to_lds_b128 %0, %1, off"
                   :: "v"(loff), "v"(ga) : "memory");
    }
  };

  Abf a0, a1;
  a0.u[0] = *(const uint4*)(qrow + 16 * hi);
  a0.u[1] = *(const uint4*)(qrow + 16 * hi + 8);
  a1.u[0] = *(const uint4*)(qrow + 32 + 16 * hi);
  a1.u[1] = *(const uint4*)(qrow + 32 + 16 * hi + 8);

  float srow[8], mv[8], rl[8];
  for (int r = 0; r < 8; ++r) {
    const int i = it * 16 + r + 8 * hi;
    srow[r] = s[(size_t)(i * 8 + b_) * 8 + h];
    mv[r]   = mrow[(size_t)bh * QLEN + i];
    rl[r]   = 1.0f / lrow[(size_t)bh * QLEN + i];
  }

  v8f acc[4] = {zero8(), zero8(), zero8(), zero8()};

  issue_async_v(0, 0);
  for (int j0 = 0; j0 < QLEN; j0 += 32) {
    const int buf = (j0 >> 5) & 1;
    const bool more = (j0 + 32) < QLEN;
    if (more) issue_async_v(j0 + 32, buf ^ 1);

    for (int half = 0; half < 2; ++half) {
      const int jt = (j0 >> 4) + half;
      const bf16_t* krow = kbase + (size_t)(jt * 16 + lo) * DHEAD;
      Abf b0, b1;
      b0.u[0] = *(const uint4*)(krow + 16 * hi);
      b0.u[1] = *(const uint4*)(krow + 16 * hi + 8);
      b1.u[0] = *(const uint4*)(krow + 32 + 16 * hi);
      b1.u[1] = *(const uint4*)(krow + 32 + 16 * hi + 8);
      v8f c = zero8();
      c = wmma_bf16(a0, b0, c);
      c = wmma_bf16(a1, b1, c);

      const int j = jt * 16 + lo;
      for (int r = 0; r < 8; ++r) {
        const int i  = it * 16 + r + 8 * hi;
        const float ac = c[r];
        const float bd = (j <= i) ? rvec[j - i + (QLEN - 1)] * srow[r] : 0.f;
        const float sc = (ac + bd) * SCALE;
        const float p  = __expf(sc - mv[r]) * rl[r];
        outP[((size_t)i * QLEN + j) * (BSZ * NHEAD) + bh] = p;
        ldsP[(r + 8 * hi) * 40 + half * 16 + lo] = f2b(p);
      }
    }
    __syncthreads();

    // group k complete when <= 8 async ops (group k+1) remain outstanding;
    // async loads complete in order (ISA 08 section 4.1).
    if (more) asm volatile("s_wait_asynccnt 0x8" ::: "memory");
    else      asm volatile("s_wait_asynccnt 0x0" ::: "memory");

    Abf pa;
    pa.u[0] = *(const uint4*)(&ldsP[lo * 40 + 16 * hi]);
    pa.u[1] = *(const uint4*)(&ldsP[lo * 40 + 16 * hi + 8]);
    for (int dt = 0; dt < 4; ++dt) {
      Abf vb;
      vb.u[0] = *(const uint4*)(&ldsV[buf][(dt * 16 + lo) * 40 + 16 * hi]);
      vb.u[1] = *(const uint4*)(&ldsV[buf][(dt * 16 + lo) * 40 + 16 * hi + 8]);
      acc[dt] = wmma_bf16(pa, vb, acc[dt]);
    }
    __syncthreads();
  }

  for (int dt = 0; dt < 4; ++dt)
    for (int r = 0; r < 8; ++r) {
      const int i = it * 16 + r + 8 * hi;
      AV[(size_t)(i * 8 + b_) * (NHEAD * DHEAD) + h * DHEAD + dt * 16 + lo] = f2b(acc[dt][r]);
    }
}

// ---------------------------------------------------------------------------
// K5: out1 = w + attn_vec @ o_W^T   ([8192x512] x [512x512])
// ---------------------------------------------------------------------------
__global__ __launch_bounds__(32) void k_outproj(const bf16_t* __restrict__ AV,
                                                const bf16_t* __restrict__ oWb,
                                                const float*  __restrict__ w,
                                                float* __restrict__ out1) {
  const int ti = blockIdx.x;   // 0..511
  const int tj = blockIdx.y;   // 0..31
  const int lane = threadIdx.x, lo = lane & 15, hi = lane >> 4;

  const bf16_t* arow = AV  + (size_t)(ti * 16 + lo) * DMODEL;
  const bf16_t* brow = oWb + (size_t)(tj * 16 + lo) * DMODEL;

  v8f c = zero8();
  for (int k0 = 0; k0 < DMODEL; k0 += 32) {
    Abf a, b;
    a.u[0] = *(const uint4*)(arow + k0 + 16 * hi);
    a.u[1] = *(const uint4*)(arow + k0 + 16 * hi + 8);
    b.u[0] = *(const uint4*)(brow + k0 + 16 * hi);
    b.u[1] = *(const uint4*)(brow + k0 + 16 * hi + 8);
    c = wmma_bf16(a, b, c);
  }

  const int gcol = tj * 16 + lo;
  for (int r = 0; r < 8; ++r) {
    const int grow = ti * 16 + r + 8 * hi;
    const size_t o = (size_t)grow * DMODEL + gcol;
    out1[o] = w[o] + c[r];
  }
}

// ---------------------------------------------------------------------------
extern "C" void kernel_launch(void* const* d_in, const int* in_sizes, int n_in,
                              void* d_out, int out_size, void* d_ws, size_t ws_size,
                              hipStream_t stream) {
  const float* w     = (const float*)d_in[0];   // [1024, 8, 512]
  const float* rvec  = (const float*)d_in[1];   // [1024, 1]
  const float* qkvW  = (const float*)d_in[2];   // [1536, 512]
  const float* rW    = (const float*)d_in[3];   // [512, 1]
  const float* oW    = (const float*)d_in[4];   // [512, 512]
  const float* rrb   = (const float*)d_in[5];   // [8, 64]
  const float* rwb   = (const float*)d_in[6];   // [8, 64]

  float* out  = (float*)d_out;
  float* out1  = out;                                    // 4,194,304
  float* outP  = out + 4194304;                          // 67,108,864
  float* outAC = out + 4194304 + 67108864;               // 67,108,864
  float* outBD = out + 4194304 + 2 * 67108864;           // 67,108,864

  char* p = (char*)d_ws;
  bf16_t* Xbf   = (bf16_t*)(p);                 // 8,388,608 B
  bf16_t* QKVbf = (bf16_t*)(p + 8388608);       // 1,572,864 B
  bf16_t* oWbf  = (bf16_t*)(p + 9961472);       //   524,288 B
  bf16_t* Qw    = (bf16_t*)(p + 10485760);      // 8,388,608 B
  bf16_t* Kst   = (bf16_t*)(p + 18874368);      // 8,388,608 B
  bf16_t* Vt    = (bf16_t*)(p + 27262976);      // 8,388,608 B
  bf16_t* AVbf  = (bf16_t*)(p + 35651584);      // 8,388,608 B
  float*  sbuf  = (float*) (p + 44040192);      //   262,144 B
  float*  mrow  = (float*) (p + 44302336);      //   262,144 B
  float*  lrow  = (float*) (p + 44564480);      //   262,144 B  (end 44,826,624)

  // K0: bf16 staging
  {
    const int total = ROWS * DMODEL + HDIM * DMODEL + DMODEL * DMODEL;
    k_convert<<<(total + 255) / 256, 256, 0, stream>>>(w, qkvW, oW, Xbf, QKVbf, oWbf);
  }
  // K1: QKV projection
  k_qkv_gemm<<<dim3(ROWS / 16, HDIM / 16), 32, 0, stream>>>(Xbf, QKVbf, rwb, Qw, Kst, Vt);
  // K2: BD rank-1 coefficients
  k_s<<<(ROWS * NHEAD + 255) / 256, 256, 0, stream>>>(Qw, rW, rrb, rwb, sbuf);
  // K3: AC/BD + softmax stats
  k_score<<<dim3(QLEN / 16, BSZ * NHEAD), 32, 0, stream>>>(Qw, Kst, rvec, sbuf,
                                                           outAC, outBD, mrow, lrow);
  // K4: prob + attn_vec (async V staging)
  k_prob_av<<<dim3(QLEN / 16, BSZ * NHEAD), 32, 0, stream>>>(Qw, Kst, Vt, rvec, sbuf,
                                                             mrow, lrow, outP, AVbf);
  // K5: output projection + residual
  k_outproj<<<dim3(ROWS / 16, DMODEL / 16), 32, 0, stream>>>(AVbf, oWbf, w, out1);
}
